// __RelativePositionalMultiheadAttentionTF_73117523247465
// MI455X (gfx1250) — matrix operand order repaired and checked by reading, not compiled, not run on hardware
//
#include <hip/hip_runtime.h>
#include <hip/hip_bf16.h>

// ---------------------------------------------------------------------------
// MI455X (gfx1250, wave32) implementation.
// All heavy math runs on v_wmma_f32_16x16x32_bf16 (fp32 accumulate).
// ---------------------------------------------------------------------------

typedef __bf16 bf16_t;
typedef __bf16 bf8   __attribute__((ext_vector_type(8)));
typedef __bf16 v16bf __attribute__((ext_vector_type(16)));
typedef float  v8f   __attribute__((ext_vector_type(8)));

// Combine two 8-element (16B) chunks into one WMMA 16-element operand.
static __device__ __forceinline__ v16bf ld8p(const bf16_t* plo, const bf16_t* phi) {
  bf8 lo = *(const bf8*)plo;
  bf8 hi = *(const bf8*)phi;
  return __builtin_shufflevector(lo, hi, 0,1,2,3,4,5,6,7,8,9,10,11,12,13,14,15);
}
static __device__ __forceinline__ v16bf ld16(const bf16_t* p) { return ld8p(p, p + 8); }

// ---------------------------------------------------------------------------
// fp32 -> bf16 pack
// ---------------------------------------------------------------------------
__global__ void k_f32_to_bf16(const float* __restrict__ src, bf16_t* __restrict__ dst, int n) {
  int i = blockIdx.x * blockDim.x + threadIdx.x;
  if (i < n) dst[i] = (bf16_t)src[i];
}

// ---------------------------------------------------------------------------
// Generic bf16 GEMM:  C[M,N] = A[M,K] @ B[K,N] + bias   (fp32 accumulate)
// Block tile 128x128, K-step 32 (one WMMA K). 256 threads = 8 waves (2x4),
// each wave owns a 64x32 sub-tile = 4x2 WMMA accumulators.
// A staged row-major in LDS; B staged COLUMN-major so that the per-lane
// B fragment (col = lane&15, k = (lane>>4)*16 .. +15) is contiguous.
//
// WMMA bf16 fragment layouts (ISA 7.12.2):
//  A: row = lane&15; elems 0-15 -> k = (lane>>4)*16 + 0..15 (this code's k order, not the ISA table's)
//  B: col = lane&15; elems 0-15 -> k = (lane>>4)*16 + 0..15 (same order as A, not the ISA table's)
//  C: row = (lane>>4)*8 + vgpr;  col = lane&15
// ---------------------------------------------------------------------------
template <bool OUT_BF16>
__global__ __launch_bounds__(256)
void k_gemm_bf16(const bf16_t* __restrict__ A, const bf16_t* __restrict__ Bm,
                 const float* __restrict__ bias, void* __restrict__ Cout,
                 int M, int N, int K) {
  __shared__ bf16_t As[128 * 32];   // [row][k]
  __shared__ bf16_t Bs[128 * 32];   // [col][k]  (column-major stage)

  const int tid  = threadIdx.x;
  const int lane = tid & 31;
  const int wave = tid >> 5;
  const int wm0  = (wave >> 2) * 64;   // wave M origin in tile
  const int wn0  = (wave & 3) * 32;    // wave N origin in tile
  const int row0 = blockIdx.y * 128;
  const int col0 = blockIdx.x * 128;

  v8f acc[4][2];
#pragma unroll
  for (int mt = 0; mt < 4; ++mt)
#pragma unroll
    for (int nt = 0; nt < 2; ++nt)
#pragma unroll
      for (int i = 0; i < 8; ++i) acc[mt][nt][i] = 0.0f;

  const int nK = K >> 5;
  for (int kt = 0; kt < nK; ++kt) {
    const int k0 = kt << 5;
    // Stage A tile: 128x32, 16B per slot, 512 slots.
#pragma unroll
    for (int i = 0; i < 2; ++i) {
      int slot = tid * 2 + i;
      int r  = slot >> 2;
      int kc = (slot & 3) * 8;
      *(bf8*)(As + r * 32 + kc) = *(const bf8*)(A + (size_t)(row0 + r) * K + k0 + kc);
    }
    // Stage B tile: 32x128 global rows -> column-major LDS.
#pragma unroll
    for (int i = 0; i < 2; ++i) {
      int slot = tid * 2 + i;
      int kr = slot >> 4;          // 0..31
      int nc = (slot & 15) * 8;    // 0..120
      bf8 v = *(const bf8*)(Bm + (size_t)(k0 + kr) * N + col0 + nc);
#pragma unroll
      for (int e = 0; e < 8; ++e) Bs[(nc + e) * 32 + kr] = v[e];
    }
    // Hint next K-slab into the cache hierarchy (global_prefetch_b8).
    if (kt + 1 < nK) {
      __builtin_prefetch(A + (size_t)(row0 + (tid >> 1)) * K + k0 + 32, 0, 0);
      __builtin_prefetch(Bm + (size_t)(k0 + 32 + (tid & 31)) * N + col0, 0, 0);
    }
    __syncthreads();

    const int koff = (lane >> 4) * 16;
    v16bf af[4];
#pragma unroll
    for (int mt = 0; mt < 4; ++mt) {
      const bf16_t* p = As + (wm0 + mt * 16 + (lane & 15)) * 32;
      af[mt] = ld16(p + koff);
    }
    v16bf bfg[2];
#pragma unroll
    for (int nt = 0; nt < 2; ++nt) {
      const bf16_t* p = Bs + (wn0 + nt * 16 + (lane & 15)) * 32 + (lane >> 4) * 16;
      bfg[nt] = ld16(p);
    }
#pragma unroll
    for (int mt = 0; mt < 4; ++mt)
#pragma unroll
      for (int nt = 0; nt < 2; ++nt)
        acc[mt][nt] = __builtin_amdgcn_wmma_f32_16x16x32_bf16(
            false, af[mt], false, bfg[nt], (short)0, acc[mt][nt], false, false);
    __syncthreads();
  }

  // Epilogue: add bias, store (M, N are multiples of 128 here, no guards).
#pragma unroll
  for (int mt = 0; mt < 4; ++mt)
#pragma unroll
    for (int nt = 0; nt < 2; ++nt)
#pragma unroll
      for (int v = 0; v < 8; ++v) {
        int row = row0 + wm0 + mt * 16 + (lane >> 4) * 8 + v;
        int col = col0 + wn0 + nt * 16 + (lane & 15);
        float val = acc[mt][nt][v] + bias[col];
        if (OUT_BF16) ((bf16_t*)Cout)[(size_t)row * N + col] = (bf16_t)val;
        else          ((float*)Cout)[(size_t)row * N + col]  = val;
      }
}

// ---------------------------------------------------------------------------
// Scatter qkv matrix [8192 x 2304] into Q(scaled)/K/V^T, reproducing the
// reference's *raw* reshape of [B,N,nh,3,hd] into [B*nh, N, hd, 3]:
//   g = b*12 + (12n+h)/1024 ; m = (12n+h)%1024
//   element (e,f):  u = 3e+j with f == (u/64 + u%64) % 3 ; comp=u/64, d=u%64
//   source col = h*192 + d*3 + comp
// ---------------------------------------------------------------------------
__global__ void k_scatter_qkv(const bf16_t* __restrict__ qkv,
                              bf16_t* __restrict__ Q, bf16_t* __restrict__ K,
                              bf16_t* __restrict__ V) {
  int i = blockIdx.x * blockDim.x + threadIdx.x;
  if (i >= 96 * 1024 * 64) return;
  int e = i & 63;
  int m = (i >> 6) & 1023;
  int g = i >> 16;
  int b = g / 12, gg = g % 12;
  int t = gg * 1024 + m;
  int n = t / 12, h = t % 12;
  size_t rowbase = ((size_t)(b * 1024 + n)) * 2304 + h * 192;
#pragma unroll
  for (int j = 0; j < 3; ++j) {
    int u = 3 * e + j;
    int comp = u >> 6;
    int dd = u & 63;
    int f = (comp + dd) % 3;
    bf16_t val = qkv[rowbase + dd * 3 + comp];
    if (f == 0)       Q[(size_t)g * 65536 + m * 64 + e] = (bf16_t)((float)val * 0.125f);
    else if (f == 1)  K[(size_t)g * 65536 + m * 64 + e] = val;
    else              V[(size_t)g * 65536 + (size_t)e * 1024 + m] = val;  // V stored transposed
  }
}

// ---------------------------------------------------------------------------
// Flash attention with decomposed relative-position bias.
// Grid: (16 q-blocks, 96 batch-heads). 128 threads = 4 waves, 16 queries/wave.
// Keys streamed in 64-wide tiles through LDS.  K tile [key][d] row-major is
// exactly the column-major B fragment layout for S = Q K^T; V is pre-
// transposed globally so the [d][key] tile is the B layout for O += P V.
// ---------------------------------------------------------------------------
__global__ __launch_bounds__(128)
void k_flash_attn(const bf16_t* __restrict__ Q, const bf16_t* __restrict__ Kk,
                  const bf16_t* __restrict__ Vt, const float* __restrict__ rph,
                  const float* __restrict__ rpw, bf16_t* __restrict__ O) {
  __shared__ bf16_t Ks[64 * 64];        // [key][d]
  __shared__ bf16_t Vs[64 * 64];        // [d][key]
  __shared__ bf16_t Ps[4 * 16 * 64];    // per-wave P tile (bf16)
  __shared__ float  rhS[4 * 16 * 32];   // per-row rel-h bias
  __shared__ float  rwS[4 * 16 * 32];   // per-row rel-w bias
  __shared__ float  mS[4 * 16], lS[4 * 16], fS[4 * 16];

  const int tid  = threadIdx.x;
  const int lane = tid & 31;
  const int wave = tid >> 5;
  const int g    = blockIdx.y;
  const int q0   = blockIdx.x * 64 + wave * 16;
  const size_t gq = (size_t)g * 1024;

  // Per-query-row bias tables: rh[r][kh] = 8 * (Q_scaled . rel_pos_h[hq-kh+31])
  // (factor 8 undoes the hd^-0.5 = 0.125 baked into Q; reference uses unscaled q).
  for (int t = lane; t < 512; t += 32) {
    int r = t >> 5, kx = t & 31;
    int qg = q0 + r;
    const bf16_t* qp = Q + (gq + qg) * 64;
    const float* ph = rph + (((qg >> 5) - kx + 31)) * 64;
    const float* pw = rpw + (((qg & 31) - kx + 31)) * 64;
    float s1 = 0.f, s2 = 0.f;
    for (int d = 0; d < 64; ++d) {
      float qv = (float)qp[d];
      s1 += qv * ph[d];
      s2 += qv * pw[d];
    }
    rhS[wave * 512 + t] = 8.0f * s1;
    rwS[wave * 512 + t] = 8.0f * s2;
  }
  if (lane < 16) { mS[wave * 16 + lane] = -3.0e38f; lS[wave * 16 + lane] = 0.0f; }

  // Resident Q fragments (16 rows x 64 d -> two WMMA K-steps).
  const int koff = (lane >> 4) * 16;
  const bf16_t* qb = Q + (gq + q0 + (lane & 15)) * 64;
  v16bf qf[2];
#pragma unroll
  for (int kc = 0; kc < 2; ++kc)
    qf[kc] = ld16(qb + kc * 32 + koff);

  v8f oacc[4];
#pragma unroll
  for (int dt = 0; dt < 4; ++dt)
#pragma unroll
    for (int i = 0; i < 8; ++i) oacc[dt][i] = 0.0f;

  for (int kk0 = 0; kk0 < 1024; kk0 += 64) {
    // Stage K tile (contiguous 8KB) and V^T tile.
    const bf16_t* kg = Kk + (gq + kk0) * 64;
#pragma unroll
    for (int i = 0; i < 4; ++i) {
      int s = (tid + i * 128) * 8;
      *(bf8*)(Ks + s) = *(const bf8*)(kg + s);
    }
#pragma unroll
    for (int i = 0; i < 4; ++i) {
      int slot = tid + i * 128;
      int d = slot >> 3, kp = (slot & 7) * 8;
      *(bf8*)(Vs + d * 64 + kp) =
          *(const bf8*)(Vt + (size_t)g * 65536 + (size_t)d * 1024 + kk0 + kp);
    }
    __syncthreads();

    // S = Q K^T over this 64-key tile: 4 column tiles x 2 K-steps = 8 WMMAs.
    v8f sacc[4];
#pragma unroll
    for (int nt = 0; nt < 4; ++nt) {
#pragma unroll
      for (int i = 0; i < 8; ++i) sacc[nt][i] = 0.0f;
      const bf16_t* bp = Ks + (nt * 16 + (lane & 15)) * 64 + (lane >> 4) * 16;
#pragma unroll
      for (int kc = 0; kc < 2; ++kc)
        sacc[nt] = __builtin_amdgcn_wmma_f32_16x16x32_bf16(
            false, qf[kc], false, ld16(bp + kc * 32), (short)0, sacc[nt], false, false);
    }

    // Online softmax: row r = (lane>>4)*8 + v spans lanes (lane&15) = cols.
#pragma unroll
    for (int v = 0; v < 8; ++v) {
      int rloc = (lane >> 4) * 8 + v;
      float mold = mS[wave * 16 + rloc];
      float sv[4];
#pragma unroll
      for (int nt = 0; nt < 4; ++nt) {
        int key = kk0 + nt * 16 + (lane & 15);
        sv[nt] = sacc[nt][v] + rhS[wave * 512 + rloc * 32 + (key >> 5)]
                             + rwS[wave * 512 + rloc * 32 + (key & 31)];
      }
      float cmax = fmaxf(fmaxf(sv[0], sv[1]), fmaxf(sv[2], sv[3]));
#pragma unroll
      for (int off = 1; off < 16; off <<= 1) cmax = fmaxf(cmax, __shfl_xor(cmax, off));
      float mnew = fmaxf(mold, cmax);
      float psum = 0.f;
#pragma unroll
      for (int nt = 0; nt < 4; ++nt) {
        float p = __expf(sv[nt] - mnew);
        psum += p;
        Ps[wave * 1024 + rloc * 64 + nt * 16 + (lane & 15)] = (bf16_t)p;
      }
#pragma unroll
      for (int off = 1; off < 16; off <<= 1) psum += __shfl_xor(psum, off);
      float fac = __expf(mold - mnew);
      if ((lane & 15) == v) {
        mS[wave * 16 + rloc] = mnew;
        lS[wave * 16 + rloc] = lS[wave * 16 + rloc] * fac + psum;
        fS[wave * 16 + rloc] = fac;
      }
    }

    // Rescale accumulators by exp(m_old - m_new).
#pragma unroll
    for (int dt = 0; dt < 4; ++dt)
#pragma unroll
      for (int v = 0; v < 8; ++v)
        oacc[dt][v] *= fS[wave * 16 + (lane >> 4) * 8 + v];

    // O += P V : 4 d-tiles x 2 K-steps = 8 WMMAs.
    v16bf pf[2];
    const bf16_t* pp = Ps + wave * 1024 + (lane & 15) * 64;
#pragma unroll
    for (int kc = 0; kc < 2; ++kc)
      pf[kc] = ld16(pp + kc * 32 + koff);
#pragma unroll
    for (int dt = 0; dt < 4; ++dt) {
      const bf16_t* vp = Vs + (dt * 16 + (lane & 15)) * 64 + (lane >> 4) * 16;
#pragma unroll
      for (int kc = 0; kc < 2; ++kc)
        oacc[dt] = __builtin_amdgcn_wmma_f32_16x16x32_bf16(
            false, pf[kc], false, ld16(vp + kc * 32), (short)0, oacc[dt], false, false);
    }
    __syncthreads();
  }

  // Final normalize + store O (bf16, [g][m][d] layout).
#pragma unroll
  for (int dt = 0; dt < 4; ++dt)
#pragma unroll
    for (int v = 0; v < 8; ++v) {
      int rloc = (lane >> 4) * 8 + v;
      float inv = 1.0f / lS[wave * 16 + rloc];
      O[(gq + q0 + rloc) * 64 + dt * 16 + (lane & 15)] = (bf16_t)(oacc[dt][v] * inv);
    }
}

// ---------------------------------------------------------------------------
// Inverse output scramble: attention output (scrambled [g,m,e] space) ->
// proj-GEMM input rows (b, H', W') x 768 features.
// ---------------------------------------------------------------------------
__global__ void k_pack_proj(const bf16_t* __restrict__ Oa, bf16_t* __restrict__ A2) {
  int i = blockIdx.x * blockDim.x + threadIdx.x;
  if (i >= 8192 * 768) return;
  int ee = i % 768;
  int row2 = i / 768;
  int b = row2 >> 10;
  int hw = row2 & 1023;
  int Hp = hw >> 5, Wp = hw & 31;
  int ft = Hp * 24576 + Wp * 768 + ee;    // flat in [nh,H,W,hd] within b
  int h2 = ft >> 16;
  int hh = (ft >> 11) & 31;
  int ww = (ft >> 6) & 31;
  int e2 = ft & 63;
  int fw = hh * 24576 + ww * 768 + h2 * 64 + e2;  // flat in scrambled [gg,m,e]
  A2[i] = Oa[(size_t)b * 786432 + fw];
}

// ---------------------------------------------------------------------------
// Launch: workspace layout (bytes; total needed ~92.8 MB)
//   xb      @        0  (12.6 MB)   Qs @ 55050240 (12.6 MB)
//   wqkvb   @ 12582912  ( 3.5 MB)   K  @ 67633152 (12.6 MB)
//   wprojb  @ 16121856  ( 1.2 MB)   Vt @ 80216064 (12.6 MB)
//   qkvb    @ 17301504  (37.7 MB)  -- reused for Oattn (front) + A2 (tail)
// ---------------------------------------------------------------------------
extern "C" void kernel_launch(void* const* d_in, const int* in_sizes, int n_in,
                              void* d_out, int out_size, void* d_ws, size_t ws_size,
                              hipStream_t stream) {
  const float* x     = (const float*)d_in[0];
  const float* Wqkv  = (const float*)d_in[1];
  const float* bqkv  = (const float*)d_in[2];
  const float* Wproj = (const float*)d_in[3];
  const float* bproj = (const float*)d_in[4];
  const float* rph   = (const float*)d_in[5];
  const float* rpw   = (const float*)d_in[6];
  float* out = (float*)d_out;

  char* ws = (char*)d_ws;
  bf16_t* xb     = (bf16_t*)(ws);
  bf16_t* wqkvb  = (bf16_t*)(ws + 12582912);
  bf16_t* wprojb = (bf16_t*)(ws + 16121856);
  bf16_t* qkvb   = (bf16_t*)(ws + 17301504);
  bf16_t* Qs     = (bf16_t*)(ws + 55050240);
  bf16_t* Kb     = (bf16_t*)(ws + 67633152);
  bf16_t* Vt     = (bf16_t*)(ws + 80216064);
  bf16_t* Oa     = qkvb;                               // reuse after scatter
  bf16_t* A2     = (bf16_t*)(ws + 17301504 + 12582912);

  k_f32_to_bf16<<<(8192 * 768 + 255) / 256, 256, 0, stream>>>(x, xb, 8192 * 768);
  k_f32_to_bf16<<<(768 * 2304 + 255) / 256, 256, 0, stream>>>(Wqkv, wqkvb, 768 * 2304);
  k_f32_to_bf16<<<(768 * 768 + 255) / 256, 256, 0, stream>>>(Wproj, wprojb, 768 * 768);

  // QKV projection: [8192 x 768] @ [768 x 2304] + bqkv -> bf16
  k_gemm_bf16<true><<<dim3(2304 / 128, 8192 / 128), 256, 0, stream>>>(
      xb, wqkvb, bqkv, qkvb, 8192, 2304, 768);

  k_scatter_qkv<<<(96 * 1024 * 64 + 255) / 256, 256, 0, stream>>>(qkvb, Qs, Kb, Vt);

  k_flash_attn<<<dim3(16, 96), 128, 0, stream>>>(Qs, Kb, Vt, rph, rpw, Oa);

  k_pack_proj<<<(8192 * 768 + 255) / 256, 256, 0, stream>>>(Oa, A2);

  // Output projection: [8192 x 768] @ [768 x 768] + bproj -> fp32 d_out
  k_gemm_bf16<false><<<dim3(768 / 128, 8192 / 128), 256, 0, stream>>>(
      A2, wprojb, bproj, out, 8192, 768, 768);
}
